// ChebNet_58076547776809
// MI455X (gfx1250) — matrix operand order repaired and checked by reading, not compiled, not run on hardware
//
#include <hip/hip_runtime.h>

// CDNA5 WMMA vector types (wave32)
typedef __attribute__((ext_vector_type(16))) __bf16 v16bf;
typedef __attribute__((ext_vector_type(8)))  float  v8f;

// ---------------------------------------------------------------------------
// Utility kernels
// ---------------------------------------------------------------------------
__global__ void zero_kernel(float* p, long n) {
    long i = (long)blockIdx.x * blockDim.x + threadIdx.x;
    if (i < n) p[i] = 0.0f;
}

__global__ void deg_kernel(const long long* __restrict__ row, int E,
                           float* __restrict__ deg) {
    int e = blockIdx.x * blockDim.x + threadIdx.x;
    if (e < E) atomicAdd(&deg[(int)row[e]], 1.0f);
}

__global__ void dinv_kernel(float* d, int N) {
    int i = blockIdx.x * blockDim.x + threadIdx.x;
    if (i < N) {
        float v = d[i];
        d[i] = (v > 0.0f) ? rsqrtf(v) : 0.0f;
    }
}

// dst[i] -= a[i]
__global__ void sub_kernel(float* __restrict__ dst, const float* __restrict__ a,
                           long n) {
    long i = (long)blockIdx.x * blockDim.x + threadIdx.x;
    if (i < n) dst[i] -= a[i];
}

// ---------------------------------------------------------------------------
// SpMM: dst[row] += factor * (-dinv[row]*dinv[col]) * src[col]
// One wave32 per edge; lanes stride the C=50 channels (coalesced row access).
// Scatter via global_atomic_add_f32: 20MB targets resolve in the 192MB L2.
// ---------------------------------------------------------------------------
__global__ void spmm_kernel(const long long* __restrict__ ei, int E,
                            const float* __restrict__ dinv,
                            const float* __restrict__ src,
                            float* __restrict__ dst, float factor) {
    constexpr int C = 50;
    int wid  = (int)(((long)blockIdx.x * blockDim.x + threadIdx.x) >> 5);
    int lane = threadIdx.x & 31;
    if (wid >= E) return;
    int r = (int)ei[wid];
    int c = (int)ei[(size_t)E + wid];
    float w = -factor * dinv[r] * dinv[c];
    const float* s = src + (size_t)c * C;
    float*       d = dst + (size_t)r * C;
    // lanes 0..31 cover ch, lanes 0..17 additionally cover ch+32
    float v0 = w * s[lane];
    if (lane < C - 32) {
        float v1 = w * s[lane + 32];
        atomicAdd(&d[lane + 32], v1);
    }
    atomicAdd(&d[lane], v0);
}

// ---------------------------------------------------------------------------
// A-fragment loader (16-bit A 16x32 layout, ISA 7.12.2). Cin = 50.
// Each lane's fragment = two contiguous 8-float runs of its row:
//   t=0: arow[hi8..hi8+7]       | arow[16+hi8..+7]     (all < 50: unconditional)
//   t=1: arow[32+hi8..+7]       | arow[48+hi8..+7]     (second run straddles 50)
// Branch-free: clamp addresses, zero pad via VALU select.
// ---------------------------------------------------------------------------
__device__ __forceinline__ void load_a_frags(const float* __restrict__ arow,
                                             int hi8, v16bf& a0, v16bf& a1) {
    const float2* p0 = reinterpret_cast<const float2*>(arow + hi8);
    const float2* p1 = reinterpret_cast<const float2*>(arow + 16 + hi8);
    const float2* p2 = reinterpret_cast<const float2*>(arow + 32 + hi8);
#pragma unroll
    for (int i = 0; i < 4; ++i) {
        float2 v = p0[i];
        a0[2 * i]     = (__bf16)v.x;
        a0[2 * i + 1] = (__bf16)v.y;
        float2 u = p1[i];
        a0[8 + 2 * i]     = (__bf16)u.x;
        a0[8 + 2 * i + 1] = (__bf16)u.y;
        float2 t = p2[i];
        a1[2 * i]     = (__bf16)t.x;
        a1[2 * i + 1] = (__bf16)t.y;
    }
#pragma unroll
    for (int i = 0; i < 8; ++i) {
        int kk = 48 + hi8 + i;                 // 48..63; valid iff < 50
        float v = arow[kk < 50 ? kk : 49];     // clamped, unconditional load
        a1[8 + i] = (__bf16)(kk < 50 ? v : 0.0f);
    }
}

// ---------------------------------------------------------------------------
// B-fragment loader (16-bit B 32x16 layout). Lane holds column `col`,
// K = (hi?8:0)+e+(e&8) for t=0 and +32 for t=1 (straddles Cin=50).
// ---------------------------------------------------------------------------
template <int Cout>
__device__ __forceinline__ void load_b_frags(const float* __restrict__ wseg,
                                             int hi16, int col,
                                             v16bf& b0, v16bf& b1) {
#pragma unroll
    for (int e = 0; e < 16; ++e) {
        b0[e] = (__bf16)wseg[(size_t)((hi16 >> 1) + e + (e & 8)) * Cout + col];  // kk <= 31 < 50
    }
#pragma unroll
    for (int e = 0; e < 16; ++e) {
        int kk = 32 + (hi16 >> 1) + e + (e & 8);                // 32..63
        float v = wseg[(size_t)(kk < 50 ? kk : 49) * Cout + col];
        b1[e] = (__bf16)(kk < 50 ? v : 0.0f);
    }
}

// ---------------------------------------------------------------------------
// Fused Chebyshev GEMM:  out = act( X0@W[0] + X1@W[1] + X2@W[2] + bias )
// One wave32 per 16x16 output tile; 6 static v_wmma_f32_16x16x32_bf16.
// ---------------------------------------------------------------------------
template <int Cout, bool RELU>
__global__ void gemm3_wmma(const float* __restrict__ X0,
                           const float* __restrict__ X1,
                           const float* __restrict__ X2,
                           const float* __restrict__ W,   // [3, 50, Cout]
                           const float* __restrict__ bias,
                           float* __restrict__ out, int N) {
    constexpr int Cin = 50;
    const int lane = threadIdx.x & 31;
    const int wid  = blockIdx.x * (blockDim.x >> 5) + (threadIdx.x >> 5);
    constexpr int colTiles = (Cout + 15) >> 4;
    const int rowTiles = (N + 15) >> 4;
    if (wid >= rowTiles * colTiles) return;  // wave-uniform: EXEC stays all-1s
    const int row0 = (wid / colTiles) << 4;
    const int col0 = (wid % colTiles) << 4;
    const int m    = lane & 15;
    const int hi   = (lane >> 4) & 1;
    const int hi8  = hi << 3;
    const int hi16 = hi << 4;

    // Clamped row/col: loads always in-bounds; OOB results never stored.
    const int rowA = min(row0 + m, N - 1);
    const int col  = min(col0 + (lane & 15), Cout - 1);

    v8f acc = {};
    v16bf a0, a1, b0, b1;

    // source 0
    load_a_frags(X0 + (size_t)rowA * Cin, hi8, a0, a1);
    load_b_frags<Cout>(W + 0 * (size_t)Cin * Cout, hi16, col, b0, b1);
    acc = __builtin_amdgcn_wmma_f32_16x16x32_bf16(false, a0, false, b0, (short)0, acc, false, false);
    acc = __builtin_amdgcn_wmma_f32_16x16x32_bf16(false, a1, false, b1, (short)0, acc, false, false);
    // source 1
    load_a_frags(X1 + (size_t)rowA * Cin, hi8, a0, a1);
    load_b_frags<Cout>(W + 1 * (size_t)Cin * Cout, hi16, col, b0, b1);
    acc = __builtin_amdgcn_wmma_f32_16x16x32_bf16(false, a0, false, b0, (short)0, acc, false, false);
    acc = __builtin_amdgcn_wmma_f32_16x16x32_bf16(false, a1, false, b1, (short)0, acc, false, false);
    // source 2
    load_a_frags(X2 + (size_t)rowA * Cin, hi8, a0, a1);
    load_b_frags<Cout>(W + 2 * (size_t)Cin * Cout, hi16, col, b0, b1);
    acc = __builtin_amdgcn_wmma_f32_16x16x32_bf16(false, a0, false, b0, (short)0, acc, false, false);
    acc = __builtin_amdgcn_wmma_f32_16x16x32_bf16(false, a1, false, b1, (short)0, acc, false, false);

    // Epilogue: C/D layout — VGPR r holds row (hi?8:0)+r, col = lane%16
    const int ncol = col0 + (lane & 15);
    if (ncol < Cout) {
        float bv = bias[ncol];
#pragma unroll
        for (int r = 0; r < 8; ++r) {
            int mrow = row0 + hi8 + r;
            if (mrow < N) {
                float v = acc[r] + bv;
                if (RELU) v = fmaxf(v, 0.0f);
                out[(size_t)mrow * Cout + ncol] = v;
            }
        }
    }
}

// ---------------------------------------------------------------------------
// Host launcher
// ---------------------------------------------------------------------------
extern "C" void kernel_launch(void* const* d_in, const int* in_sizes, int n_in,
                              void* d_out, int out_size, void* d_ws, size_t ws_size,
                              hipStream_t stream) {
    (void)n_in; (void)out_size; (void)ws_size;
    const float*     x   = (const float*)d_in[0];
    const long long* ei  = (const long long*)d_in[1];  // int64 [2,E]
    const float*     W1  = (const float*)d_in[2];      // [3,50,50]
    const float*     b1  = (const float*)d_in[3];
    const float*     W2  = (const float*)d_in[4];      // [3,50,40]
    const float*     b2  = (const float*)d_in[5];
    float*           out = (float*)d_out;

    const int CIN = 50, CH = 50, COUT = 40;
    const int N = in_sizes[0] / CIN;
    const int E = in_sizes[1] / 2;

    // Workspace (floats): dinv[N] | TxA[N*CH] | TxB[N*CH] | h[N*CH]
    float* dinv = (float*)d_ws;
    float* TxA  = dinv + N;
    float* TxB  = TxA + (size_t)N * CH;
    float* h    = TxB + (size_t)N * CH;

    const long NC = (long)N * CH;
    const int TB = 256;
    dim3 blk(TB);
    dim3 gN((N + TB - 1) / TB);
    dim3 gE((E + TB - 1) / TB);
    dim3 gNC((unsigned)((NC + TB - 1) / TB));
    dim3 gEw((unsigned)(((long)E * 32 + TB - 1) / TB));  // wave per edge

    // Degree -> dinv (in place)
    zero_kernel<<<gN, blk, 0, stream>>>(dinv, N);
    deg_kernel<<<gE, blk, 0, stream>>>(ei, E, dinv);
    dinv_kernel<<<gN, blk, 0, stream>>>(dinv, N);

    auto gemm_grid = [&](int cout) {
        int tiles = ((N + 15) >> 4) * ((cout + 15) >> 4);
        return dim3((unsigned)((tiles + 3) / 4));  // 4 waves / 128-thread block
    };
    dim3 gblk(128);

    // ---- Layer 1: Tx0 = x, Tx1 = L@x, Tx2 = 2 L@Tx1 - x ----
    zero_kernel<<<gNC, blk, 0, stream>>>(TxA, NC);
    spmm_kernel<<<gEw, blk, 0, stream>>>(ei, E, dinv, x, TxA, 1.0f);
    zero_kernel<<<gNC, blk, 0, stream>>>(TxB, NC);
    spmm_kernel<<<gEw, blk, 0, stream>>>(ei, E, dinv, TxA, TxB, 2.0f);
    sub_kernel<<<gNC, blk, 0, stream>>>(TxB, x, NC);
    gemm3_wmma<50, true><<<gemm_grid(CH), gblk, 0, stream>>>(x, TxA, TxB, W1, b1, h, N);

    // ---- Layer 2: Tx0 = h ----
    zero_kernel<<<gNC, blk, 0, stream>>>(TxA, NC);
    spmm_kernel<<<gEw, blk, 0, stream>>>(ei, E, dinv, h, TxA, 1.0f);
    zero_kernel<<<gNC, blk, 0, stream>>>(TxB, NC);
    spmm_kernel<<<gEw, blk, 0, stream>>>(ei, E, dinv, TxA, TxB, 2.0f);
    sub_kernel<<<gNC, blk, 0, stream>>>(TxB, h, NC);
    gemm3_wmma<40, false><<<gemm_grid(COUT), gblk, 0, stream>>>(h, TxA, TxB, W2, b2, out, N);
}
